// TabM_packed_34703335751965
// MI455X (gfx1250) — matrix operand order repaired and checked by reading, not compiled, not run on hardware
//
#include <hip/hip_runtime.h>
#include <hip/hip_bf16.h>

typedef __bf16 bf16_t;
typedef bf16_t v16bf __attribute__((ext_vector_type(16)));
typedef float  v8f   __attribute__((ext_vector_type(8)));

#define ENS_K  32
#define DIM_B  4096
#define DIM_F  512
#define DIM_H  1024
#define DIM_C  100
#define DIM_CP 112
#define N_HID  3

// GEMM tiling
#define BM 128
#define BN 128
#define BK 32
#define LDA 56   // padded LDS row stride in halfs: 112B (16B aligned, bank-conflict-free for 16B frag reads)

union FragAB { v16bf v; uint4 q[2]; };

// Raw LDS byte offset of a __shared__ pointer (LDS aperture: addr[31:0] = LDS offset)
#define LDS_OFF(p) ((unsigned)(size_t)(p))

// CDNA5 async global->LDS copy, tracked by ASYNCcnt (bypasses VGPRs).
__device__ __forceinline__ void async_ld_b128(unsigned ldsOff, const void* gsrc) {
  asm volatile("global_load_async_to_lds_b128 %0, %1, off"
               :: "v"(ldsOff), "v"(gsrc) : "memory");
}
__device__ __forceinline__ void wait_async0() {
  asm volatile("s_wait_asynccnt 0" ::: "memory");
}

// ---------------------------------------------------------------------------
// fp32 -> bf16 elementwise convert
// ---------------------------------------------------------------------------
__global__ void __launch_bounds__(256) convert_f32_bf16(const float* __restrict__ src,
                                                        bf16_t* __restrict__ dst, int n) {
  int i = blockIdx.x * blockDim.x + threadIdx.x;
  if (i < n) dst[i] = (bf16_t)src[i];
}

// ---------------------------------------------------------------------------
// Batched transpose + convert: src [R, Cc] fp32 -> dst [CcPad, R] bf16,
// zero-filling padded rows. Standard 32x32 LDS tile transpose.
// ---------------------------------------------------------------------------
__global__ void __launch_bounds__(256) transpose_convert(const float* __restrict__ src,
                                                         bf16_t* __restrict__ dst,
                                                         int R, int Cc, int CcPad,
                                                         long sStride, long dStride) {
  __shared__ float tile[32][33];
  const long b = blockIdx.z;
  const int c0 = blockIdx.x * 32, r0 = blockIdx.y * 32;
  const int tx = threadIdx.x, ty = threadIdx.y;   // block (32, 8)
  #pragma unroll
  for (int ii = 0; ii < 4; ++ii) {
    int r = r0 + ty + ii * 8, c = c0 + tx;
    tile[ty + ii * 8][tx] = (r < R && c < Cc) ? src[b * sStride + (long)r * Cc + c] : 0.0f;
  }
  __syncthreads();
  #pragma unroll
  for (int ii = 0; ii < 4; ++ii) {
    int dr = c0 + ty + ii * 8;   // output row = source column
    int dc = r0 + tx;            // output col = source row
    if (dr < CcPad && dc < R)
      dst[b * dStride + (long)dr * R + dc] = (bf16_t)tile[tx][ty + ii * 8];
  }
}

// ---------------------------------------------------------------------------
// Batched-ensemble bf16 GEMM:  out[k] = relu?(act[k] @ wt[k]^T + bias[k])
// 128x128x32 tiles, 8 waves (2x4), wave tile 64x32 (4x2 WMMA 16x16x32 tiles).
// Async global->LDS staging with double buffering: one barrier per K-step.
// ---------------------------------------------------------------------------
__global__ void __launch_bounds__(256) gemm_bf16(
    const bf16_t* __restrict__ act, long actKStride,
    const bf16_t* __restrict__ wt,  long wtKStride,
    const float*  __restrict__ bias, long biasKStride,
    bf16_t* __restrict__ out, long outKStride,
    int N, int KD, int doRelu) {
  __shared__ __align__(16) bf16_t As[2][BM * LDA];
  __shared__ __align__(16) bf16_t Bs[2][BN * LDA];

  const int t    = threadIdx.x;
  const int lane = t & 31;
  const int w    = t >> 5;
  const int wm   = (w >> 2) * 64;   // wave row origin within block tile
  const int wn   = (w & 3) * 32;    // wave col origin within block tile
  const int l15  = lane & 15;
  const int rsel = lane >> 4;

  const long kz = blockIdx.z;
  const bf16_t* actBase = act + kz * actKStride + (long)(blockIdx.y * BM) * KD;
  const bf16_t* wtBase  = wt  + kz * wtKStride  + (long)(blockIdx.x * BN) * KD;

  // Per-thread staging slots: 512 uint4 per tile, 2 per thread.
  const int r0s = t >> 2;                 // rows 0..63
  const int r1s = r0s + 64;               // rows 64..127
  const int cs  = (t & 3) * 8;            // half-column chunk

  auto issue = [&](int buf, int kt) {
    async_ld_b128(LDS_OFF(&As[buf][r0s * LDA + cs]), actBase + (long)r0s * KD + kt + cs);
    async_ld_b128(LDS_OFF(&As[buf][r1s * LDA + cs]), actBase + (long)r1s * KD + kt + cs);
    async_ld_b128(LDS_OFF(&Bs[buf][r0s * LDA + cs]), wtBase + (long)r0s * KD + kt + cs);
    async_ld_b128(LDS_OFF(&Bs[buf][r1s * LDA + cs]), wtBase + (long)r1s * KD + kt + cs);
  };

  const v8f vzero = {0.f, 0.f, 0.f, 0.f, 0.f, 0.f, 0.f, 0.f};
  v8f acc[4][2];
  #pragma unroll
  for (int i = 0; i < 4; ++i)
    #pragma unroll
    for (int j = 0; j < 2; ++j) acc[i][j] = vzero;

  issue(0, 0);
  const int nsteps = KD / BK;
  for (int it = 0; it < nsteps; ++it) {
    const int cur = it & 1;
    wait_async0();        // own async writes to As/Bs[cur] complete
    __syncthreads();      // everyone's writes visible; previous reads of [1-cur] done
    if (it + 1 < nsteps) issue(1 - cur, (it + 1) * BK);

    FragAB a[4], b[2];
    // A frag: row M = l15; lanes 0-15 hold K[0:8)+K[16:24), lanes 16-31 K[8:16)+K[24:32)
    #pragma unroll
    for (int i = 0; i < 4; ++i) {
      const bf16_t* p = &As[cur][(wm + i * 16 + l15) * LDA + rsel * 8];
      a[i].q[0] = *(const uint4*)p;
      a[i].q[1] = *(const uint4*)(p + 16);
    }
    // B frag: col N = l15; lanes 0-15 hold K[0:8)+K[16:24), lanes 16-31 K[8:16)+K[24:32)
    #pragma unroll
    for (int j = 0; j < 2; ++j) {
      const bf16_t* p = &Bs[cur][(wn + j * 16 + l15) * LDA + rsel * 8];
      b[j].q[0] = *(const uint4*)p;
      b[j].q[1] = *(const uint4*)(p + 16);
    }
    #pragma unroll
    for (int i = 0; i < 4; ++i)
      #pragma unroll
      for (int j = 0; j < 2; ++j)
        acc[i][j] = __builtin_amdgcn_wmma_f32_16x16x32_bf16(
            false, a[i].v, false, b[j].v, (short)0, acc[i][j], false, false);
  }

  // Epilogue: bias + relu + bf16 store. C/D layout: row = v + 8*rsel, col = l15.
  const float* biasK = bias + kz * biasKStride;
  bf16_t* outK = out + kz * outKStride;
  #pragma unroll
  for (int j = 0; j < 2; ++j) {
    int col = blockIdx.x * BN + wn + j * 16 + l15;
    float bv = biasK[col];
    #pragma unroll
    for (int i = 0; i < 4; ++i) {
      int row0 = blockIdx.y * BM + wm + i * 16 + rsel * 8;
      #pragma unroll
      for (int v = 0; v < 8; ++v) {
        float r = acc[i][j][v] + bv;
        if (doRelu) r = fmaxf(r, 0.0f);
        outK[(long)(row0 + v) * N + col] = (bf16_t)r;
      }
    }
  }
}

// ---------------------------------------------------------------------------
// Output layer + mean over ensembles:
//   out[b,c] = (1/K) * sum_k ( h[k,b,:] @ Wout[k,:,c] + bout[k,c] )
// Block = 128 rows x 112 (padded) cols; 8 waves, wave = 16 rows x 7 n-tiles.
// Flattened 1024-step (k x kt) loop, async double-buffered staging.
// ---------------------------------------------------------------------------
__global__ void __launch_bounds__(256) out_mean(
    const bf16_t* __restrict__ h, const bf16_t* __restrict__ wt,
    const float* __restrict__ bout, float* __restrict__ outp) {
  __shared__ __align__(16) bf16_t As[2][BM * LDA];
  __shared__ __align__(16) bf16_t Bs[2][DIM_CP * LDA];

  const int t = threadIdx.x;
  const int lane = t & 31;
  const int w = t >> 5;
  const int l15 = lane & 15;
  const int rsel = lane >> 4;
  const long rowBase = (long)blockIdx.x * BM;

  const int r0s = t >> 2;        // rows 0..63
  const int r1s = r0s + 64;      // rows 64..127 (Bs: valid only up to 111 -> t < 192)
  const int cs  = (t & 3) * 8;

  auto issue = [&](int buf, int s) {  // s = flattened step: k = s>>5, kt = (s&31)*BK
    const int k = s >> 5;
    const int kt = (s & 31) * BK;
    const bf16_t* hK  = h  + (long)k * DIM_B * DIM_H + rowBase * DIM_H + kt;
    const bf16_t* wtK = wt + (long)k * DIM_CP * DIM_H + kt;
    async_ld_b128(LDS_OFF(&As[buf][r0s * LDA + cs]), hK + (long)r0s * DIM_H + cs);
    async_ld_b128(LDS_OFF(&As[buf][r1s * LDA + cs]), hK + (long)r1s * DIM_H + cs);
    async_ld_b128(LDS_OFF(&Bs[buf][r0s * LDA + cs]), wtK + (long)r0s * DIM_H + cs);
    if (t < 192)  // wave-uniform: waves 0-5 stage Bs rows 64..111
      async_ld_b128(LDS_OFF(&Bs[buf][r1s * LDA + cs]), wtK + (long)r1s * DIM_H + cs);
  };

  const v8f vzero = {0.f, 0.f, 0.f, 0.f, 0.f, 0.f, 0.f, 0.f};
  v8f acc[7];
  #pragma unroll
  for (int j = 0; j < 7; ++j) acc[j] = vzero;

  const int nsteps = ENS_K * (DIM_H / BK);  // 1024
  issue(0, 0);
  for (int s = 0; s < nsteps; ++s) {
    const int cur = s & 1;
    wait_async0();
    __syncthreads();
    if (s + 1 < nsteps) issue(1 - cur, s + 1);

    FragAB a;
    const bf16_t* pa = &As[cur][(w * 16 + l15) * LDA + rsel * 8];
    a.q[0] = *(const uint4*)pa;
    a.q[1] = *(const uint4*)(pa + 16);

    #pragma unroll
    for (int j = 0; j < 7; ++j) {
      FragAB b;
      const bf16_t* pb = &Bs[cur][(j * 16 + l15) * LDA + rsel * 8];
      b.q[0] = *(const uint4*)pb;
      b.q[1] = *(const uint4*)(pb + 16);
      acc[j] = __builtin_amdgcn_wmma_f32_16x16x32_bf16(
          false, a.v, false, b.v, (short)0, acc[j], false, false);
    }
  }

  const float invK = 1.0f / (float)ENS_K;
  #pragma unroll
  for (int j = 0; j < 7; ++j) {
    int col = j * 16 + l15;
    if (col < DIM_C) {
      float bsum = 0.0f;
      for (int k = 0; k < ENS_K; ++k) bsum += bout[k * DIM_C + col];
      long row0 = rowBase + w * 16 + rsel * 8;
      #pragma unroll
      for (int v = 0; v < 8; ++v)
        outp[(row0 + v) * DIM_C + col] = (acc[j][v] + bsum) * invK;
    }
  }
}

// ---------------------------------------------------------------------------
extern "C" void kernel_launch(void* const* d_in, const int* in_sizes, int n_in,
                              void* d_out, int out_size, void* d_ws, size_t ws_size,
                              hipStream_t stream) {
  (void)in_sizes; (void)n_in; (void)out_size; (void)ws_size;

  const float* x     = (const float*)d_in[0];
  const float* W_in  = (const float*)d_in[1];
  const float* b_in  = (const float*)d_in[2];
  const float* W_hid = (const float*)d_in[3];
  const float* b_hid = (const float*)d_in[4];
  const float* W_out = (const float*)d_in[5];
  const float* b_out = (const float*)d_in[6];
  float* outp = (float*)d_out;

  // Workspace layout (bf16 buffers)
  char* ws = (char*)d_ws;
  size_t off = 0;
  bf16_t* xb    = (bf16_t*)(ws + off); off += (size_t)DIM_B * DIM_F * 2;              // 4 MiB
  bf16_t* wtin  = (bf16_t*)(ws + off); off += (size_t)ENS_K * DIM_H * DIM_F * 2;      // 32 MiB [k][H][F]
  bf16_t* wthid = (bf16_t*)(ws + off); off += (size_t)N_HID * ENS_K * DIM_H * DIM_H * 2; // 192 MiB
  bf16_t* wtout = (bf16_t*)(ws + off); off += (size_t)ENS_K * DIM_CP * DIM_H * 2;     // 7 MiB [k][112][H]
  bf16_t* hA    = (bf16_t*)(ws + off); off += (size_t)ENS_K * DIM_B * DIM_H * 2;      // 256 MiB
  bf16_t* hB    = (bf16_t*)(ws + off);                                                // 256 MiB

  // 1) Convert x to bf16
  convert_f32_bf16<<<(DIM_B * DIM_F + 255) / 256, 256, 0, stream>>>(x, xb, DIM_B * DIM_F);

  // 2) Transpose-convert weights to bf16 [N][K] layout
  transpose_convert<<<dim3(DIM_H / 32, DIM_F / 32, ENS_K), dim3(32, 8), 0, stream>>>(
      W_in, wtin, DIM_F, DIM_H, DIM_H, (long)DIM_F * DIM_H, (long)DIM_H * DIM_F);
  transpose_convert<<<dim3(DIM_H / 32, DIM_H / 32, N_HID * ENS_K), dim3(32, 8), 0, stream>>>(
      W_hid, wthid, DIM_H, DIM_H, DIM_H, (long)DIM_H * DIM_H, (long)DIM_H * DIM_H);
  transpose_convert<<<dim3(4, DIM_H / 32, ENS_K), dim3(32, 8), 0, stream>>>(
      W_out, wtout, DIM_H, DIM_C, DIM_CP, (long)DIM_H * DIM_C, (long)DIM_CP * DIM_H);

  // 3) Input layer: h = relu(x @ W_in[k] + b_in[k])   (x shared across k)
  dim3 ggrid(DIM_H / BN, DIM_B / BM, ENS_K);
  gemm_bf16<<<ggrid, 256, 0, stream>>>(xb, 0L,
                                       wtin, (long)DIM_H * DIM_F,
                                       b_in, (long)DIM_H,
                                       hA, (long)DIM_B * DIM_H,
                                       DIM_H, DIM_F, 1);

  // 4) Hidden layers (ping-pong hA/hB)
  bf16_t* bufs[2] = {hA, hB};
  for (int i = 0; i < N_HID; ++i) {
    const bf16_t* srcb = bufs[i & 1];
    bf16_t* dstb = bufs[(i + 1) & 1];
    gemm_bf16<<<ggrid, 256, 0, stream>>>(srcb, (long)DIM_B * DIM_H,
                                         wthid + (size_t)i * ENS_K * DIM_H * DIM_H,
                                         (long)DIM_H * DIM_H,
                                         b_hid + (size_t)i * ENS_K * DIM_H, (long)DIM_H,
                                         dstb, (long)DIM_B * DIM_H,
                                         DIM_H, DIM_H, 1);
  }
  const bf16_t* hFinal = bufs[N_HID & 1];  // N_HID=3 -> hB

  // 5) Output layer + mean over ensembles
  out_mean<<<DIM_B / BM, 256, 0, stream>>>(hFinal, wtout, b_out, outp);
}
